// DividedAttentionSublayer_84404697301733
// MI455X (gfx1250) — matrix operand order repaired and checked by reading, not compiled, not run on hardware
//
#include <hip/hip_runtime.h>

// ---------------- problem constants ----------------
#define B_    8
#define L_    1024
#define D_    1024
#define H_    16
#define DH_   64
#define NROW  (B_ * L_)       // 8192
#define MAXPOS 128
#define RPOS  257             // 2*MAXPOS + 1
#define RPAD  272             // RPOS padded to multiple of 16
#define NEG_BIG (-1.0e30f)

typedef __attribute__((ext_vector_type(16))) __bf16 v16bf;
typedef __attribute__((ext_vector_type(8)))  __bf16 v8bf;
typedef __attribute__((ext_vector_type(8)))  float  v8f;
typedef __attribute__((ext_vector_type(4)))  unsigned int u32x4;
typedef __attribute__((ext_vector_type(8)))  int i32x8;
typedef __attribute__((ext_vector_type(4)))  int i32x4;

#if defined(__HIP_DEVICE_COMPILE__) && __has_builtin(__builtin_amdgcn_tensor_load_to_lds) && __has_builtin(__builtin_amdgcn_s_wait_tensorcnt)
#define USE_TDM 1
#else
#define USE_TDM 0
#endif

// ---------------- workspace layout (bytes) ----------------
#define OFF_WQ   ((size_t)0)
#define OFF_WK   ((size_t)2097152)
#define OFF_WV   ((size_t)4194304)
#define OFF_WO   ((size_t)6291456)
#define OFF_POS  ((size_t)8388608)                 // RPAD*DH bf16, zero padded
#define OFF_XQ   ((size_t)(8388608 + 65536))       // [B*L, D] bf16
#define OFF_XK   (OFF_XQ + (size_t)16777216)
#define OFF_XV   (OFF_XK + (size_t)16777216)
#define OFF_Q    (OFF_XV + (size_t)16777216)       // [B,H,L,DH] bf16 (pre-scaled 1/8)
#define OFF_K    (OFF_Q  + (size_t)16777216)       // [B,H,L,DH] bf16
#define OFF_VT   (OFF_K  + (size_t)16777216)       // [B,H,DH,L] bf16 (transposed)
#define OFF_HB   (OFF_VT + (size_t)16777216)       // [B*L, D] bf16 heads-concat

__device__ __forceinline__ v16bf cat8(v8bf lo, v8bf hi) {
  return __builtin_shufflevector(lo, hi, 0,1,2,3,4,5,6,7,8,9,10,11,12,13,14,15);
}
__device__ __forceinline__ v8f wmma_bf16(v16bf a, v16bf b, v8f c) {
  return __builtin_amdgcn_wmma_f32_16x16x32_bf16(false, a, false, b, (short)0, c,
                                                 false, false);
}
__device__ __forceinline__ v8f vzero8() {
  v8f z = {0.f,0.f,0.f,0.f,0.f,0.f,0.f,0.f};
  return z;
}

#if USE_TDM
// Tensor Data Mover: 2-D tile load global->LDS (bf16, data_size=1 -> 2 bytes).
// D# per CDNA5 ISA.  All args wave-uniform.
// tile_d0 = contiguous elements per row, tile_d1 = rows,
// stride0 = row stride of the source tensor in elements.
// This toolchain's builtin is the 6-arg variant:
//   (u32x4 g0, i32x8 g1, i32x4 g2, i32x4 g3, i32x8 g1b, i32 cpol)
__device__ __forceinline__ void tdm_load_2d(unsigned lds_addr, const void* gaddr,
                                            unsigned tile_d0, unsigned tile_d1,
                                            unsigned tensor_d0, unsigned tensor_d1,
                                            unsigned stride0) {
  unsigned long long ga = (unsigned long long)(size_t)gaddr;
  u32x4 g0;
  g0[0] = 1u;                                            // count=1 (user, valid)
  g0[1] = lds_addr;                                      // lds byte address
  g0[2] = (unsigned)(ga & 0xffffffffu);                  // global_addr[31:0]
  g0[3] = (unsigned)((ga >> 32) & 0x01ffffffu) | (2u << 30);  // addr[56:32]|type=2
  i32x8 g1;
  g1[0] = (int)(1u << 16);                               // data_size=1 (2B)
  g1[1] = (int)((tensor_d0 & 0xffffu) << 16);            // tensor_dim0[15:0]
  g1[2] = (int)((tensor_d0 >> 16) | ((tensor_d1 & 0xffffu) << 16));
  g1[3] = (int)((tensor_d1 >> 16) | (tile_d0 << 16));    // tile_dim0
  g1[4] = (int)(tile_d1 & 0xffffu);                      // tile_dim1, tile_dim2=0
  g1[5] = (int)stride0;                                  // dim0_stride[31:0]
  g1[6] = 0;                                             // dim0_stride[47:32]=0
  g1[7] = 0;
  i32x4 gz4 = {0, 0, 0, 0};
  i32x8 gz8 = {0, 0, 0, 0, 0, 0, 0, 0};
  __builtin_amdgcn_tensor_load_to_lds(g0, g1, gz4, gz4, gz8, 0);
}
__device__ __forceinline__ unsigned lds_off(const void* p) {
  return (unsigned)(size_t)p;   // low 32 bits of generic LDS pointer = LDS offset
}
#endif

// =======================================================================
// Kernel 1: fp32 -> bf16 conversions: weights, pos_emb (zero padded),
//           and the three activation tensors.
// =======================================================================
__global__ void prep_kernel(const float* __restrict__ Wq, const float* __restrict__ Wk,
                            const float* __restrict__ Wv, const float* __restrict__ Wo,
                            const float* __restrict__ pos,
                            const float* __restrict__ xq, const float* __restrict__ xk,
                            const float* __restrict__ xv,
                            __bf16* __restrict__ wqb, __bf16* __restrict__ wkb,
                            __bf16* __restrict__ wvb, __bf16* __restrict__ wob,
                            __bf16* __restrict__ posb,
                            __bf16* __restrict__ xqb, __bf16* __restrict__ xkb,
                            __bf16* __restrict__ xvb) {
  const size_t i = (size_t)blockIdx.x * blockDim.x + threadIdx.x;
  if (i < (size_t)D_ * D_) {
    wqb[i] = (__bf16)Wq[i];
    wkb[i] = (__bf16)Wk[i];
    wvb[i] = (__bf16)Wv[i];
    wob[i] = (__bf16)Wo[i];
  }
  if (i < (size_t)RPAD * DH_) {
    int r = (int)(i / DH_);
    posb[i] = (r < RPOS) ? (__bf16)pos[i] : (__bf16)0.0f;
  }
  if (i < (size_t)NROW * D_) {
    xqb[i] = (__bf16)xq[i];
    xkb[i] = (__bf16)xk[i];
    xvb[i] = (__bf16)xv[i];
  }
}

// =======================================================================
// 64x64-per-wave WMMA GEMM body:  out = A @ W^T + bias
//   mode 0: Q  (scale 0.125) -> [B,H,L,DH] bf16
//   mode 1: K                -> [B,H,L,DH] bf16
//   mode 2: V  (transposed)  -> [B,H,DH,L] bf16
//   mode 3: fp32 row-major [NROW, D]
// =======================================================================
__device__ __forceinline__
void gemm64_body(const __bf16* __restrict__ A, const __bf16* __restrict__ W,
                 const float* __restrict__ bias, void* __restrict__ outp,
                 const int mode) {
  const int lane = threadIdx.x;
  const int m  = lane & 15;
  const int hi = lane >> 4;
  const int r0 = blockIdx.x * 64;
  const int c0 = blockIdx.y * 64;

  v8f acc[4][4];
#pragma unroll
  for (int ra = 0; ra < 4; ++ra)
#pragma unroll
    for (int t = 0; t < 4; ++t) acc[ra][t] = vzero8();

  for (int kc = 0; kc < D_ / 32; ++kc) {
    const int kb = kc * 32;
    // stream-ahead prefetch (global_prefetch_b8), 256 elements ahead
    if (kc + 8 < D_ / 32) {
      __builtin_prefetch(A + (size_t)(r0 + lane) * D_ + kb + 256, 0, 1);
      __builtin_prefetch(A + (size_t)(r0 + 32 + lane) * D_ + kb + 256, 0, 1);
      __builtin_prefetch(W + (size_t)(c0 + lane) * D_ + kb + 256, 0, 1);
      __builtin_prefetch(W + (size_t)(c0 + 32 + lane) * D_ + kb + 256, 0, 1);
    }
    // ---- 4 A fragments (rows r0+ra*16+m), khalf = hi*16
    v16bf a[4];
#pragma unroll
    for (int ra = 0; ra < 4; ++ra) {
      const __bf16* ar = A + (size_t)(r0 + ra * 16 + m) * D_ + kb;
      a[ra] = cat8(*(const v8bf*)(ar + hi * 16), *(const v8bf*)(ar + hi * 16 + 8));
    }
    // ---- 4 B fragments: b[j] = W[c0+t*16+(l&15)][kb + hi*16 + j]
    v16bf bt[4];
#pragma unroll
    for (int t = 0; t < 4; ++t) {
      const __bf16* wp = W + (size_t)(c0 + t * 16 + m) * D_ + kb + hi * 16;
      bt[t] = cat8(*(const v8bf*)wp, *(const v8bf*)(wp + 8));
    }
    // ---- 16 WMMAs
#pragma unroll
    for (int ra = 0; ra < 4; ++ra)
#pragma unroll
      for (int t = 0; t < 4; ++t) acc[ra][t] = wmma_bf16(a[ra], bt[t], acc[ra][t]);
  }

  // ---- epilogue: C layout lane l -> (M = hi*8+v, N = l&15)
#pragma unroll
  for (int t = 0; t < 4; ++t) {
    const int col = c0 + t * 16 + m;
    const float bi = bias[col];
    const int hh = col >> 6, dd = col & 63;
#pragma unroll
    for (int ra = 0; ra < 4; ++ra) {
#pragma unroll
      for (int v = 0; v < 8; ++v) {
        const int grow = r0 + ra * 16 + hi * 8 + v;
        const int bb = grow >> 10, ll = grow & 1023;
        const float val = acc[ra][t][v] + bi;
        if (mode == 0) {
          ((__bf16*)outp)[((size_t)(bb * H_ + hh) * L_ + ll) * DH_ + dd] =
              (__bf16)(val * 0.125f);
        } else if (mode == 1) {
          ((__bf16*)outp)[((size_t)(bb * H_ + hh) * L_ + ll) * DH_ + dd] = (__bf16)val;
        } else if (mode == 2) {
          ((__bf16*)outp)[((size_t)(bb * H_ + hh) * DH_ + dd) * L_ + ll] = (__bf16)val;
        } else {
          ((float*)outp)[(size_t)grow * D_ + col] = val;
        }
      }
    }
  }
}

// Fused QKV: gridDim.z = 3 selects the matrix -> 6144 waves in one launch.
__global__ __launch_bounds__(32)
void qkv_gemm_kernel(const __bf16* __restrict__ xqb, const __bf16* __restrict__ xkb,
                     const __bf16* __restrict__ xvb,
                     const __bf16* __restrict__ wqb, const __bf16* __restrict__ wkb,
                     const __bf16* __restrict__ wvb,
                     const float* __restrict__ bq, const float* __restrict__ bk,
                     const float* __restrict__ bv,
                     __bf16* __restrict__ Qb, __bf16* __restrict__ Kb,
                     __bf16* __restrict__ Vt) {
  const int z = blockIdx.z;
  const __bf16* A = (z == 0) ? xqb : (z == 1) ? xkb : xvb;
  const __bf16* W = (z == 0) ? wqb : (z == 1) ? wkb : wvb;
  const float* bias = (z == 0) ? bq : (z == 1) ? bk : bv;
  void* outp = (z == 0) ? (void*)Qb : (z == 1) ? (void*)Kb : (void*)Vt;
  gemm64_body(A, W, bias, outp, z);
}

__global__ __launch_bounds__(32)
void oproj_gemm_kernel(const __bf16* __restrict__ Hb, const __bf16* __restrict__ wob,
                       const float* __restrict__ bo, float* __restrict__ out) {
  gemm64_body(Hb, wob, bo, (void*)out, 3);
}

// =======================================================================
// Kernel 3: flash attention with relative-position logits.
//   One wave per (b,h, 32-row q-tile).  Qpos = Qtile @ pos_embT in LDS,
//   gathered as S[q,k] += Qpos[q][clamp(k-q)+128].  K blocks of 32,
//   K/V tiles double-buffered via the Tensor Data Mover when available.
// =======================================================================
__global__ __launch_bounds__(32)
void attn_kernel(const __bf16* __restrict__ Qb, const __bf16* __restrict__ Kb,
                 const __bf16* __restrict__ Vt, const __bf16* __restrict__ posb,
                 const unsigned char* __restrict__ mask,
                 __bf16* __restrict__ Hb) {
  __shared__ __bf16 Qs[32][64];           // q-tile (pre-scaled by 1/8)
  __shared__ float  Qpos[32][RPAD];       // q-tile x pos logits
  __shared__ __bf16 Ksb[2][32][64];       // K block [k][d], double buffered
  __shared__ __bf16 Vsb[2][64][32];       // V block transposed [d][k], dbl buf
  __shared__ __bf16 Ps[32][32];           // P tile for C->A layout conversion
  __shared__ unsigned char Ms[32];        // mask bytes

  const int lane = threadIdx.x;
  const int m  = lane & 15;
  const int hi = lane >> 4;
  const int q0 = blockIdx.x * 32;
  const int bh = blockIdx.y;
  const int b  = bh >> 4;

  // ---- load Q tile (4096 B contiguous)
  {
    const uint4* s = (const uint4*)(Qb + ((size_t)bh * L_ + q0) * DH_);
    uint4* d = (uint4*)&Qs[0][0];
#pragma unroll
    for (int i = 0; i < 8; ++i) d[lane + 32 * i] = s[lane + 32 * i];
  }
  __syncthreads();

  // ---- A fragments of Q (2 row-halves x 2 K-chunks), reused everywhere
  v16bf aq[2][2];
#pragma unroll
  for (int qh = 0; qh < 2; ++qh)
#pragma unroll
    for (int c = 0; c < 2; ++c) {
      v8bf lo = *(const v8bf*)&Qs[qh * 16 + m][c * 32 + hi * 16];
      v8bf h8 = *(const v8bf*)&Qs[qh * 16 + m][c * 32 + hi * 16 + 8];
      aq[qh][c] = cat8(lo, h8);
    }

  const __bf16* ksrc = Kb + (size_t)bh * L_ * DH_;
  const __bf16* vsrc = Vt + (size_t)bh * DH_ * L_;
  const unsigned char* msrc = mask + (size_t)b * L_;

#if USE_TDM
  // kick off block 0 K/V tile DMAs; they overlap the Qpos GEMM below
  tdm_load_2d(lds_off(&Ksb[0][0][0]), ksrc, DH_, 32, DH_, L_, DH_);
  tdm_load_2d(lds_off(&Vsb[0][0][0]), vsrc, 32, DH_, L_, DH_, L_);
#endif

  // ---- Qpos[q][r] = Q[q]·pos_emb[r]  (17 N-tiles of 16, B frags shared)
  for (int rt = 0; rt < RPAD / 16; ++rt) {
    v16bf bt[2];
#pragma unroll
    for (int c = 0; c < 2; ++c) {
      const __bf16* pp = posb + (size_t)(rt * 16 + m) * DH_ + c * 32 + hi * 16;
      bt[c] = cat8(*(const v8bf*)pp, *(const v8bf*)(pp + 8));
    }
#pragma unroll
    for (int qh = 0; qh < 2; ++qh) {
      v8f acc = vzero8();
      acc = wmma_bf16(aq[qh][0], bt[0], acc);
      acc = wmma_bf16(aq[qh][1], bt[1], acc);
#pragma unroll
      for (int v = 0; v < 8; ++v) Qpos[qh * 16 + hi * 8 + v][rt * 16 + m] = acc[v];
    }
  }
  __syncthreads();

  // ---- online softmax state (rows M = qh*16 + hi*8 + v, replicated x16)
  float rm[2][8], rl[2][8];
  v8f o[2][4];
#pragma unroll
  for (int qh = 0; qh < 2; ++qh) {
#pragma unroll
    for (int v = 0; v < 8; ++v) { rm[qh][v] = NEG_BIG; rl[qh][v] = 0.f; }
#pragma unroll
    for (int nt = 0; nt < 4; ++nt) o[qh][nt] = vzero8();
  }

  for (int kblk = 0; kblk < L_ / 32; ++kblk) {
    const int k0 = kblk * 32;
    int cur;
#if USE_TDM
    cur = kblk & 1;
    if (kblk + 1 < L_ / 32) {
      const int kn = k0 + 32;
      tdm_load_2d(lds_off(&Ksb[cur ^ 1][0][0]), ksrc + (size_t)kn * DH_,
                  DH_, 32, DH_, L_, DH_);
      tdm_load_2d(lds_off(&Vsb[cur ^ 1][0][0]), vsrc + kn,
                  32, DH_, L_, DH_, L_);
    }
    Ms[lane] = msrc[k0 + lane];
    if (kblk + 1 < L_ / 32) __builtin_amdgcn_s_wait_tensorcnt(2);
    else                    __builtin_amdgcn_s_wait_tensorcnt(0);
    __syncthreads();
#else
    cur = 0;
    __syncthreads();
    {
      const uint4* s = (const uint4*)(ksrc + (size_t)k0 * DH_);
      uint4* d = (uint4*)&Ksb[0][0][0];
#pragma unroll
      for (int i = 0; i < 8; ++i) d[lane + 32 * i] = s[lane + 32 * i];
    }
    {
#pragma unroll
      for (int i = 0; i < 8; ++i) {
        const int idx = lane + 32 * i;        // 0..255
        const int row = idx >> 2, ch = idx & 3;
        ((uint4*)&Vsb[0][row][0])[ch] =
            ((const uint4*)(vsrc + (size_t)row * L_ + k0))[ch];
      }
    }
    Ms[lane] = msrc[k0 + lane];
    if (kblk + 1 < L_ / 32) {
      __builtin_prefetch(ksrc + (size_t)(k0 + 32) * DH_ + (size_t)lane * DH_, 0, 1);
      __builtin_prefetch(vsrc + (size_t)lane * L_ + k0 + 32, 0, 1);
      __builtin_prefetch(vsrc + (size_t)(lane + 32) * L_ + k0 + 32, 0, 1);
    }
    __syncthreads();
#endif
    __bf16 (*Kc)[64] = Ksb[cur];
    __bf16 (*Vc)[32] = Vsb[cur];

    // ---- S = Q Kt (+pos, +mask): 2 k-tiles x 2 q-halves
    float sv[2][2][8];
#pragma unroll
    for (int t = 0; t < 2; ++t) {
      v16bf btk[2];
#pragma unroll
      for (int c = 0; c < 2; ++c) {
        const __bf16* kp = &Kc[t * 16 + m][c * 32 + hi * 16];
        btk[c] = cat8(*(const v8bf*)kp, *(const v8bf*)(kp + 8));
      }
#pragma unroll
      for (int qh = 0; qh < 2; ++qh) {
        v8f acc = vzero8();
        acc = wmma_bf16(aq[qh][0], btk[0], acc);
        acc = wmma_bf16(aq[qh][1], btk[1], acc);
#pragma unroll
        for (int v = 0; v < 8; ++v) {
          const int kk = k0 + t * 16 + m;
          const int q  = q0 + qh * 16 + hi * 8 + v;
          int rel = kk - q;
          rel = rel < -MAXPOS ? -MAXPOS : (rel > MAXPOS ? MAXPOS : rel);
          float s = acc[v] + Qpos[qh * 16 + hi * 8 + v][rel + MAXPOS];
          if (Ms[t * 16 + m]) s = NEG_BIG;
          sv[qh][t][v] = s;
        }
      }
    }

    // ---- online softmax update + P tile to LDS (bf16)
#pragma unroll
    for (int qh = 0; qh < 2; ++qh) {
#pragma unroll
      for (int v = 0; v < 8; ++v) {
        float rmax = fmaxf(sv[qh][0][v], sv[qh][1][v]);
        rmax = fmaxf(rmax, __shfl_xor(rmax, 1));
        rmax = fmaxf(rmax, __shfl_xor(rmax, 2));
        rmax = fmaxf(rmax, __shfl_xor(rmax, 4));
        rmax = fmaxf(rmax, __shfl_xor(rmax, 8));
        const float mnew  = fmaxf(rm[qh][v], rmax);
        const float alpha = __expf(rm[qh][v] - mnew);
        const float p0 = __expf(sv[qh][0][v] - mnew);
        const float p1 = __expf(sv[qh][1][v] - mnew);
        float rsum = p0 + p1;
        rsum += __shfl_xor(rsum, 1);
        rsum += __shfl_xor(rsum, 2);
        rsum += __shfl_xor(rsum, 4);
        rsum += __shfl_xor(rsum, 8);
        rl[qh][v] = rl[qh][v] * alpha + rsum;
        rm[qh][v] = mnew;
        o[qh][0][v] *= alpha; o[qh][1][v] *= alpha;
        o[qh][2][v] *= alpha; o[qh][3][v] *= alpha;
        Ps[qh * 16 + hi * 8 + v][m]      = (__bf16)p0;
        Ps[qh * 16 + hi * 8 + v][16 + m] = (__bf16)p1;
      }
    }
    __syncthreads();

    // ---- O += P @ V   (A = P 16x32 per q-half, B = V d-tiles, shared)
    v16bf ap[2];
#pragma unroll
    for (int qh = 0; qh < 2; ++qh) {
      v8bf lo = *(const v8bf*)&Ps[qh * 16 + m][hi * 16];
      v8bf h8 = *(const v8bf*)&Ps[qh * 16 + m][hi * 16 + 8];
      ap[qh] = cat8(lo, h8);
    }
#pragma unroll
    for (int nt = 0; nt < 4; ++nt) {
      const __bf16* vp = &Vc[nt * 16 + m][hi * 16];
      v16bf bt = cat8(*(const v8bf*)vp, *(const v8bf*)(vp + 8));
      o[0][nt] = wmma_bf16(ap[0], bt, o[0][nt]);
      o[1][nt] = wmma_bf16(ap[1], bt, o[1][nt]);
    }
  }

  // ---- normalize + store heads-concat [b][q][h*64+d] as bf16
  __bf16* hdst = Hb + ((size_t)b * L_ + q0) * D_ + (size_t)(bh & 15) * DH_;
#pragma unroll
  for (int qh = 0; qh < 2; ++qh) {
#pragma unroll
    for (int v = 0; v < 8; ++v) {
      const float inv = 1.0f / rl[qh][v];
      const int row = qh * 16 + hi * 8 + v;
#pragma unroll
      for (int nt = 0; nt < 4; ++nt) {
        hdst[(size_t)row * D_ + nt * 16 + m] = (__bf16)(o[qh][nt][v] * inv);
      }
    }
  }
}

// =======================================================================
extern "C" void kernel_launch(void* const* d_in, const int* in_sizes, int n_in,
                              void* d_out, int out_size, void* d_ws, size_t ws_size,
                              hipStream_t stream) {
  const float* xq  = (const float*)d_in[0];
  const float* xk  = (const float*)d_in[1];
  const float* xv  = (const float*)d_in[2];
  const unsigned char* mask = (const unsigned char*)d_in[3];
  const float* Wq  = (const float*)d_in[4];
  const float* bq  = (const float*)d_in[5];
  const float* Wk  = (const float*)d_in[6];
  const float* bk  = (const float*)d_in[7];
  const float* Wv  = (const float*)d_in[8];
  const float* bv  = (const float*)d_in[9];
  const float* Wo  = (const float*)d_in[10];
  const float* bo  = (const float*)d_in[11];
  const float* pos = (const float*)d_in[12];

  char* ws = (char*)d_ws;
  __bf16* wqb  = (__bf16*)(ws + OFF_WQ);
  __bf16* wkb  = (__bf16*)(ws + OFF_WK);
  __bf16* wvb  = (__bf16*)(ws + OFF_WV);
  __bf16* wob  = (__bf16*)(ws + OFF_WO);
  __bf16* posb = (__bf16*)(ws + OFF_POS);
  __bf16* xqb  = (__bf16*)(ws + OFF_XQ);
  __bf16* xkb  = (__bf16*)(ws + OFF_XK);
  __bf16* xvb  = (__bf16*)(ws + OFF_XV);
  __bf16* Qb   = (__bf16*)(ws + OFF_Q);
  __bf16* Kb   = (__bf16*)(ws + OFF_K);
  __bf16* Vt   = (__bf16*)(ws + OFF_VT);
  __bf16* Hb   = (__bf16*)(ws + OFF_HB);

  prep_kernel<<<((size_t)NROW * D_ + 255) / 256, 256, 0, stream>>>(
      Wq, Wk, Wv, Wo, pos, xq, xk, xv,
      wqb, wkb, wvb, wob, posb, xqb, xkb, xvb);

  // Fused QKV projections (64x64 tile per wave, 3 matrices in one launch)
  qkv_gemm_kernel<<<dim3(NROW / 64, D_ / 64, 3), 32, 0, stream>>>(
      xqb, xkb, xvb, wqb, wkb, wvb, bq, bk, bv, Qb, Kb, Vt);

  // attention (32 q-rows per wave, TDM double-buffered K/V tiles)
  attn_kernel<<<dim3(L_ / 32, B_ * H_), 32, 0, stream>>>(
      Qb, Kb, Vt, posb, mask, Hb);

  // output projection, fp32 result
  oproj_gemm_kernel<<<dim3(NROW / 64, D_ / 64), 32, 0, stream>>>(
      Hb, wob, bo, (float*)d_out);
}
